// TMultiHeadAttention_86543591014955
// MI455X (gfx1250) — matrix operand order repaired and checked by reading, not compiled, not run on hardware
//
#include <hip/hip_runtime.h>

// ---------------------------------------------------------------------------
// TMultiHeadAttention on MI455X (gfx1250, wave32, WMMA).
//
// Pipeline (B=8, T=128, N=64, C=HD=256, H=4, dh=64, M=B*T*N=65536):
//   1) k_prep_w  : WQ/WK/WV/fcW fp32 -> f16 (one-time, tiny)
//   2) k_proj_qk : Qh/Kh[m][d] = x.W^T  (computed transposed: A=W16 rows,
//                  B columns = x rows, both contiguous; packed v8h stores)
//   3) k_proj_v  : V projection, stored pre-transposed Vt[b][n2][d][t2]
//   4) k_attn    : per (b,h,n2): S^T = K.Q^T (WMMA) -> mask/scale -> LDS,
//                  row softmax (unnormalized exp f16 + 1/sum),
//                  ctx^T = Vt.P^T (WMMA), normalize in epilogue
//   5) k_fc      : out = ctx.fcW^T + b (transposed GEMM, f32 out)
//
// All WMMA ops are V_WMMA_F32_16X16X32_F16 (f32 accumulate). The f16 path is
// chosen because the workload is bandwidth-bound (~0.44 GB moved -> ~19 us at
// 23.3 TB/s) while fp32 WMMA (16x16x4) would make it compute-bound.
// Every operand load / result store is a contiguous >=16B access by choosing
// GEMM orientation (transposed vs not) per stage.
// ---------------------------------------------------------------------------

typedef __attribute__((ext_vector_type(16))) _Float16 v16h;
typedef __attribute__((ext_vector_type(8)))  _Float16 v8h;
typedef __attribute__((ext_vector_type(4)))  _Float16 v4h;
typedef __attribute__((ext_vector_type(8)))  float    v8f;

#define WMMA16(a, b, c) \
  __builtin_amdgcn_wmma_f32_16x16x32_f16(false, (a), false, (b), (short)0, (c), false, false)

__device__ __forceinline__ v16h cat16(v8h lo, v8h hi) {
  return __builtin_shufflevector(lo, hi, 0, 1, 2, 3, 4, 5, 6, 7, 8, 9, 10, 11, 12, 13, 14, 15);
}

__device__ __forceinline__ v8h cvt8(const float* p) {
  float4 a = *(const float4*)p;
  float4 b = *(const float4*)(p + 4);
  v8h h;
  h[0] = (_Float16)a.x; h[1] = (_Float16)a.y; h[2] = (_Float16)a.z; h[3] = (_Float16)a.w;
  h[4] = (_Float16)b.x; h[5] = (_Float16)b.y; h[6] = (_Float16)b.z; h[7] = (_Float16)b.w;
  return h;
}

// A operand (16x32, f16): lane row q, K-group g holds K {g*8..g*8+7, 16+g*8..16+g*8+7}
__device__ __forceinline__ v16h tileA_h(const _Float16* src, int ld, int row, int kb, int g) {
  const _Float16* p = src + row * ld + kb + g * 8;
  return cat16(*(const v8h*)p, *(const v8h*)(p + 16));
}
__device__ __forceinline__ v16h tileA_f(const float* src, int ld, int row, int kb, int g) {
  const float* p = src + row * ld + kb + g * 8;
  return cat16(cvt8(p), cvt8(p + 16));
}
// B operand (32x16, f16): lane column q, group g holds K {g*8..g*8+7, 16+g*8..16+g*8+7}.
// Source row `row` of a row-major matrix supplies column `row` of B.
__device__ __forceinline__ v16h tileB_h(const _Float16* src, int ld, int row, int kb, int g) {
  const _Float16* p = src + row * ld + kb + g * 8;
  return cat16(*(const v8h*)p, *(const v8h*)(p + 16));
}
__device__ __forceinline__ v16h tileB_f(const float* src, int ld, int row, int kb, int g) {
  const float* p = src + row * ld + kb + g * 8;
  return cat16(cvt8(p), cvt8(p + 16));
}

// ---------------------------------------------------------------------------
// 1) weight conversion: [WQ | WK | WV | fcW] (each 256x256 f32) -> f16
// ---------------------------------------------------------------------------
__global__ void __launch_bounds__(256) k_prep_w(const float* __restrict__ WQ,
                                                const float* __restrict__ WK,
                                                const float* __restrict__ WV,
                                                const float* __restrict__ FW,
                                                _Float16* __restrict__ w16) {
  int i = blockIdx.x * 256 + threadIdx.x;   // grid = 1024 -> 262144 elements
  int sel = i >> 16;
  int j = i & 65535;
  float v = (sel == 0) ? WQ[j] : (sel == 1) ? WK[j] : (sel == 2) ? WV[j] : FW[j];
  w16[i] = (_Float16)v;
}

// ---------------------------------------------------------------------------
// 2) Q/K projection, transposed GEMM: Y[m][d] = sum_c W16[d][c] * x[m][c]
//    block: 64 m-columns x full 256 d; 8 waves = (4 d-groups x 2 m-groups),
//    wave tile 64(d) x 32(m) = 4x2 WMMA tiles. grid (1024,1,2) z->Q/K.
// ---------------------------------------------------------------------------
__global__ void __launch_bounds__(256) k_proj_qk(const float* __restrict__ xQ,
                                                 const float* __restrict__ xK,
                                                 const _Float16* __restrict__ w16,
                                                 _Float16* __restrict__ Qh,
                                                 _Float16* __restrict__ Kh) {
  const float* x = blockIdx.z ? xK : xQ;
  const _Float16* W = w16 + (blockIdx.z ? 65536 : 0);
  _Float16* Y = blockIdx.z ? Kh : Qh;
  const int tid = threadIdx.x;
  const int w = tid >> 5, lane = tid & 31;
  const int q = lane & 15, g = lane >> 4;
  const int d0 = (w & 3) * 64;
  const int m0 = blockIdx.x * 64 + (w >> 2) * 32;

  v8f acc[4][2] = {};
#pragma unroll
  for (int ks = 0; ks < 8; ++ks) {
    const int kb = ks * 32;
    v16h a[4], b[2];
#pragma unroll
    for (int jm = 0; jm < 4; ++jm) a[jm] = tileA_h(W, 256, d0 + jm * 16 + q, kb, g);
#pragma unroll
    for (int jn = 0; jn < 2; ++jn) b[jn] = tileB_f(x, 256, m0 + jn * 16 + q, kb, g);
#pragma unroll
    for (int jm = 0; jm < 4; ++jm)
#pragma unroll
      for (int jn = 0; jn < 2; ++jn) acc[jm][jn] = WMMA16(a[jm], b[jn], acc[jm][jn]);
  }
  // D tile: lane holds col n=m (q), rows d = base + g*8 + e  -> 8 consecutive d
#pragma unroll
  for (int jm = 0; jm < 4; ++jm)
#pragma unroll
    for (int jn = 0; jn < 2; ++jn) {
      const int m = m0 + jn * 16 + q;
      const int d = d0 + jm * 16 + g * 8;
      v8h o;
#pragma unroll
      for (int e = 0; e < 8; ++e) o[e] = (_Float16)acc[jm][jn][e];
      *(v8h*)(Y + m * 256 + d) = o;
    }
}

// ---------------------------------------------------------------------------
// 3) V projection, normal GEMM: y[m][d] = sum_c x[m][c] * W16[d][c],
//    stored transposed: Vt[((b*64+n2)*256 + d)*128 + t2], m = b*8192+n2*128+t2.
//    Wave tile 32(m) x 64(d) = 2x4 tiles; lane holds 8 consecutive m -> v8h.
// ---------------------------------------------------------------------------
__global__ void __launch_bounds__(256) k_proj_v(const float* __restrict__ xV,
                                                const _Float16* __restrict__ w16,
                                                _Float16* __restrict__ Vt) {
  const _Float16* W = w16 + 131072;
  const int tid = threadIdx.x;
  const int w = tid >> 5, lane = tid & 31;
  const int q = lane & 15, g = lane >> 4;
  const int d0 = (w & 3) * 64;
  const int m0 = blockIdx.x * 64 + (w >> 2) * 32;

  v8f acc[2][4] = {};
#pragma unroll
  for (int ks = 0; ks < 8; ++ks) {
    const int kb = ks * 32;
    v16h a[2], b[4];
#pragma unroll
    for (int jm = 0; jm < 2; ++jm) a[jm] = tileA_f(xV, 256, m0 + jm * 16 + q, kb, g);
#pragma unroll
    for (int jn = 0; jn < 4; ++jn) b[jn] = tileB_h(W, 256, d0 + jn * 16 + q, kb, g);
#pragma unroll
    for (int jm = 0; jm < 2; ++jm)
#pragma unroll
      for (int jn = 0; jn < 4; ++jn) acc[jm][jn] = WMMA16(a[jm], b[jn], acc[jm][jn]);
  }
#pragma unroll
  for (int jm = 0; jm < 2; ++jm)
#pragma unroll
    for (int jn = 0; jn < 4; ++jn) {
      const int d = d0 + jn * 16 + q;
      const int mc = m0 + jm * 16 + g * 8;  // 8 consecutive m = consecutive t2
      const int bn = mc >> 13;
      const int rem = mc & 8191;
      const int n2 = rem >> 7, t2 = rem & 127;
      v8h o;
#pragma unroll
      for (int e = 0; e < 8; ++e) o[e] = (_Float16)acc[jm][jn][e];
      *(v8h*)(Vt + (((bn * 64 + n2) * 256 + d) << 7) + t2) = o;
    }
}

// ---------------------------------------------------------------------------
// 4) Attention, one block per (b,h,n2). LDS: Pf[128][132] f32 scores,
//    rsum[128], Ph[128][136] f16 exp-values (padded strides: conflict-free).
// ---------------------------------------------------------------------------
#define PF_STRIDE 132
#define PH_STRIDE 136
#define RS_OFF (128 * PF_STRIDE * 4)          // 67584
#define PH_OFF (RS_OFF + 512)                 // 68096
#define ATTN_LDS (PH_OFF + 128 * PH_STRIDE * 2)  // 102912 bytes

__global__ void __launch_bounds__(256) k_attn(const int* __restrict__ mask,
                                              const _Float16* __restrict__ Qh,
                                              const _Float16* __restrict__ Kh,
                                              const _Float16* __restrict__ Vt,
                                              _Float16* __restrict__ Ctx) {
  extern __shared__ char smem[];
  float* Pf = (float*)smem;
  float* rsum = (float*)(smem + RS_OFF);
  _Float16* Ph = (_Float16*)(smem + PH_OFF);

  const int gblk = blockIdx.x;
  const int bb = gblk >> 8;
  const int hh = (gblk >> 6) & 3;
  const int n2 = gblk & 63;
  const int gm = bb * 8192 + n2 * 128;  // first row m of this group

  const int tid = threadIdx.x;
  const int w = tid >> 5, lane = tid & 31;
  const int q = lane & 15, g = lane >> 4;

  // ---- Phase 1: S^T[s][t] = sum_d K[s][d] * Q[t][d], scale + mask -> Pf[t][s]
  {
    const int s0 = w * 16;  // wave's s-strip
    const _Float16* Kg = Kh + gm * 256 + hh * 64;
    const _Float16* Qg = Qh + gm * 256 + hh * 64;
    v8f acc[8] = {};
#pragma unroll
    for (int ks = 0; ks < 2; ++ks) {
      const int kb = ks * 32;
      v16h a = tileA_h(Kg, 256, s0 + q, kb, g);
#pragma unroll
      for (int jt = 0; jt < 8; ++jt) {
        v16h bm = tileB_h(Qg, 256, jt * 16 + q, kb, g);
        acc[jt] = WMMA16(a, bm, acc[jt]);
      }
    }
    const float scale = 0.125f;  // 1/sqrt(64)
    const int sb = s0 + g * 8;
#pragma unroll
    for (int jt = 0; jt < 8; ++jt) {
      const int t = jt * 16 + q;
      const int* mr = mask + (((bb * 64 + n2) * 128 + t) * 128 + sb);
      int4 mA = *(const int4*)mr;
      int4 mB = *(const int4*)(mr + 4);
      float4 oA, oB;
      oA.x = mA.x ? -1e9f : acc[jt][0] * scale;
      oA.y = mA.y ? -1e9f : acc[jt][1] * scale;
      oA.z = mA.z ? -1e9f : acc[jt][2] * scale;
      oA.w = mA.w ? -1e9f : acc[jt][3] * scale;
      oB.x = mB.x ? -1e9f : acc[jt][4] * scale;
      oB.y = mB.y ? -1e9f : acc[jt][5] * scale;
      oB.z = mB.z ? -1e9f : acc[jt][6] * scale;
      oB.w = mB.w ? -1e9f : acc[jt][7] * scale;
      *(float4*)&Pf[t * PF_STRIDE + sb] = oA;
      *(float4*)&Pf[t * PF_STRIDE + sb + 4] = oB;
    }
  }
  __syncthreads();

  // ---- Phase 2: per-row softmax: Ph = exp(x-max) (f16), rsum = 1/sum
  if (tid < 128) {
    const float* row = &Pf[tid * PF_STRIDE];
    float mx = -3.0e38f;
#pragma unroll 8
    for (int i = 0; i < 128; i += 4) {
      float4 v = *(const float4*)&row[i];
      mx = fmaxf(mx, fmaxf(fmaxf(v.x, v.y), fmaxf(v.z, v.w)));
    }
    _Float16* hrow = Ph + tid * PH_STRIDE;
    float s = 0.f;
#pragma unroll 8
    for (int i = 0; i < 128; i += 4) {
      float4 v = *(const float4*)&row[i];
      v4h h;
      h[0] = (_Float16)__expf(v.x - mx);
      h[1] = (_Float16)__expf(v.y - mx);
      h[2] = (_Float16)__expf(v.z - mx);
      h[3] = (_Float16)__expf(v.w - mx);
      s += (float)h[0] + (float)h[1] + (float)h[2] + (float)h[3];
      *(v4h*)(hrow + i) = h;
    }
    rsum[tid] = 1.0f / s;
  }
  __syncthreads();

  // ---- Phase 3: ctx^T[d][t] = sum_s Vt[d][s] * P[t][s]; normalize; store
  {
    const int d0 = (w & 3) * 16;
    const int t0 = (w >> 2) * 64;
    const _Float16* Vg = Vt + ((bb * 64 + n2) * 256 + hh * 64 + d0) * 128;
    v8f acc[4] = {};
#pragma unroll
    for (int ks = 0; ks < 4; ++ks) {
      const int kb = ks * 32;
      v16h a = tileA_h(Vg, 128, q, kb, g);
#pragma unroll
      for (int jt = 0; jt < 4; ++jt) {
        const _Float16* pr = Ph + (t0 + jt * 16 + q) * PH_STRIDE + kb + g * 8;
        v16h bm = cat16(*(const v8h*)pr, *(const v8h*)(pr + 16));
        acc[jt] = WMMA16(a, bm, acc[jt]);
      }
    }
#pragma unroll
    for (int jt = 0; jt < 4; ++jt) {
      const int t = t0 + jt * 16 + q;
      const float inv = rsum[t];
      v8h o;
#pragma unroll
      for (int e = 0; e < 8; ++e) o[e] = (_Float16)(acc[jt][e] * inv);
      *(v8h*)(Ctx + (gm + t) * 256 + hh * 64 + d0 + g * 8) = o;
    }
  }
}

// ---------------------------------------------------------------------------
// 5) Final FC, transposed GEMM: out[m][c] = sum_d fcW16[c][d]*ctx[m][d] + b[c]
// ---------------------------------------------------------------------------
__global__ void __launch_bounds__(256) k_fc(const _Float16* __restrict__ Ctx,
                                            const _Float16* __restrict__ w16,
                                            const float* __restrict__ fcb,
                                            float* __restrict__ out) {
  const _Float16* W = w16 + 196608;
  const int tid = threadIdx.x;
  const int w = tid >> 5, lane = tid & 31;
  const int q = lane & 15, g = lane >> 4;
  const int c0w = (w & 3) * 64;
  const int m0 = blockIdx.x * 64 + (w >> 2) * 32;

  v8f acc[4][2] = {};
#pragma unroll
  for (int ks = 0; ks < 8; ++ks) {
    const int kb = ks * 32;
    v16h a[4], b[2];
#pragma unroll
    for (int jm = 0; jm < 4; ++jm) a[jm] = tileA_h(W, 256, c0w + jm * 16 + q, kb, g);
#pragma unroll
    for (int jn = 0; jn < 2; ++jn) b[jn] = tileB_h(Ctx, 256, m0 + jn * 16 + q, kb, g);
#pragma unroll
    for (int jm = 0; jm < 4; ++jm)
#pragma unroll
      for (int jn = 0; jn < 2; ++jn) acc[jm][jn] = WMMA16(a[jm], b[jn], acc[jm][jn]);
  }
#pragma unroll
  for (int jm = 0; jm < 4; ++jm)
#pragma unroll
    for (int jn = 0; jn < 2; ++jn) {
      const int m = m0 + jn * 16 + q;
      const int c0 = c0w + jm * 16 + g * 8;
      float4 b0 = *(const float4*)(fcb + c0);
      float4 b1 = *(const float4*)(fcb + c0 + 4);
      float4 o0, o1;
      o0.x = acc[jm][jn][0] + b0.x;
      o0.y = acc[jm][jn][1] + b0.y;
      o0.z = acc[jm][jn][2] + b0.z;
      o0.w = acc[jm][jn][3] + b0.w;
      o1.x = acc[jm][jn][4] + b1.x;
      o1.y = acc[jm][jn][5] + b1.y;
      o1.z = acc[jm][jn][6] + b1.z;
      o1.w = acc[jm][jn][7] + b1.w;
      *(float4*)(out + m * 256 + c0) = o0;
      *(float4*)(out + m * 256 + c0 + 4) = o1;
    }
}

// ---------------------------------------------------------------------------
// Host-side launcher. Workspace layout (f16 elements):
//   [0, 262144)            weight copies Wq16|Wk16|Wv16|fcW16
//   +4 x 16777216          Qh, Kh, Vt, Ctx   (total ~128.5 MB)
// ---------------------------------------------------------------------------
extern "C" void kernel_launch(void* const* d_in, const int* in_sizes, int n_in,
                              void* d_out, int out_size, void* d_ws, size_t ws_size,
                              hipStream_t stream) {
  (void)in_sizes; (void)n_in; (void)out_size; (void)ws_size;
  const float* inQ = (const float*)d_in[0];
  const float* inK = (const float*)d_in[1];
  const float* inV = (const float*)d_in[2];
  const int*   msk = (const int*)d_in[3];
  const float* WQ  = (const float*)d_in[4];
  const float* WK  = (const float*)d_in[5];
  const float* WV  = (const float*)d_in[6];
  const float* FW  = (const float*)d_in[7];
  const float* fcb = (const float*)d_in[8];
  float* out = (float*)d_out;

  _Float16* w16 = (_Float16*)d_ws;
  _Float16* Qh  = w16 + 262144;
  _Float16* Kh  = Qh + 16777216;
  _Float16* Vt  = Kh + 16777216;
  _Float16* Ctx = Vt + 16777216;

  k_prep_w<<<1024, 256, 0, stream>>>(WQ, WK, WV, FW, w16);
  k_proj_qk<<<dim3(1024, 1, 2), 256, 0, stream>>>(inQ, inK, w16, Qh, Kh);
  k_proj_v<<<1024, 256, 0, stream>>>(inV, w16, Vt);
  k_attn<<<2048, 256, ATTN_LDS, stream>>>(msk, Qh, Kh, Vt, Ctx);
  k_fc<<<1024, 256, 0, stream>>>(Ctx, w16, fcb, out);
}
